// DotProductAttention_23175643529346
// MI455X (gfx1250) — matrix operand order repaired and checked by reading, not compiled, not run on hardware
//
#include <hip/hip_runtime.h>
#include <hip/hip_bf16.h>
#include <math.h>

// ---------------------------------------------------------------------------
// Problem constants (from reference): B=8, TQ=TK=2048, QD=KD=ID=VD=512
// ---------------------------------------------------------------------------
constexpr int NB = 8;
constexpr int TT = 2048;
constexpr int DD = 512;
constexpr float INV_TEMP = 1.0f / 22.627416997969522f;

typedef __bf16 bf16_t;
typedef __attribute__((ext_vector_type(16))) __bf16 v16bf;
typedef __attribute__((ext_vector_type(8)))  __bf16 v8bf;
typedef __attribute__((ext_vector_type(4)))  __bf16 v4bf;
typedef __attribute__((ext_vector_type(8)))  float  v8f;
typedef __attribute__((ext_vector_type(4)))  unsigned int v4u;
typedef __attribute__((ext_vector_type(8)))  int v8i;
typedef __attribute__((ext_vector_type(4)))  int v4i;

// ---------------------------------------------------------------------------
// WMMA wrapper: D = A(16x32 bf16) * B(32x16 bf16) + C(16x16 f32)
// ---------------------------------------------------------------------------
__device__ inline v8f wmma_bf16(v16bf a, v16bf b, v8f c) {
  return __builtin_amdgcn_wmma_f32_16x16x32_bf16(
      /*neg_a=*/false, a, /*neg_b=*/false, b,
      /*c_mod=*/(short)0, c, /*reuse_a=*/false, /*reuse_b=*/false);
}

// ---------------------------------------------------------------------------
// Fragment loaders (wave32); A and B share contiguous k order k = 16*(lane>>4) + e.
// A 16x32 (16-bit): lanes 0-15 hold row M=lane, K in {0..15};
//                   lanes 16-31 hold row M=lane-16, K in {16..31}.
//   element e (0..15): k = ((lane>>4)<<4) + e
// B 32x16 (16-bit): lane holds column N=lane&15; K-range = (lane&16)+e.
// C/D f32 16x16: vgpr v holds row M = v + 8*(lane>=16), column N = lane&15.
// ---------------------------------------------------------------------------
__device__ inline v16bf load_afrag_f32(const float* __restrict__ base, int ld,
                                       int lane) {
  int m  = lane & 15;
  int kb = (lane >> 4) << 4;             // 0 or 16
  const float* p0 = base + (size_t)m * ld + kb;        // k = kb..kb+7
  const float* p1 = p0 + 8;                            // k = kb+8..kb+15
  float t[16];
  *(float4*)(t + 0)  = *(const float4*)(p0 + 0);
  *(float4*)(t + 4)  = *(const float4*)(p0 + 4);
  *(float4*)(t + 8)  = *(const float4*)(p1 + 0);
  *(float4*)(t + 12) = *(const float4*)(p1 + 4);
  v16bf r;
#pragma unroll
  for (int i = 0; i < 16; ++i) r[i] = (__bf16)t[i];
  return r;
}

__device__ inline v16bf load_afrag_bf16(const bf16_t* __restrict__ base,
                                        int ld, int lane) {
  int m  = lane & 15;
  int kb = (lane >> 4) << 4;
  const bf16_t* p = base + (size_t)m * ld + kb;
  v8bf lo = *(const v8bf*)(p);        // k = kb..kb+7
  v8bf hi = *(const v8bf*)(p + 8);    // k = kb+8..kb+15
  v16bf r;
#pragma unroll
  for (int i = 0; i < 8; ++i) { r[i] = lo[i]; r[8 + i] = hi[i]; }
  return r;
}

// B fragment when the logical B matrix is "dot of rows": source[n][k] row-major
// (x·Wᵀ projections, attn·Vᵀ-transposed). Per-lane contiguous.
__device__ inline v16bf load_bfrag_rows_f32(const float* __restrict__ base,
                                            int ld, int lane) {
  int n    = lane & 15;
  int koff = lane & 16;
  const float* p = base + (size_t)n * ld + koff;
  float t[16];
  *(float4*)(t + 0)  = *(const float4*)(p + 0);
  *(float4*)(t + 4)  = *(const float4*)(p + 4);
  *(float4*)(t + 8)  = *(const float4*)(p + 8);
  *(float4*)(t + 12) = *(const float4*)(p + 12);
  v16bf r;
#pragma unroll
  for (int i = 0; i < 16; ++i) r[i] = (__bf16)t[i];
  return r;
}

__device__ inline v16bf load_bfrag_rows_bf16(const bf16_t* __restrict__ base,
                                             int ld, int lane) {
  int n    = lane & 15;
  int koff = lane & 16;
  return *(const v16bf*)(base + (size_t)n * ld + koff);   // 32B aligned
}

// ---------------------------------------------------------------------------
// Tensor Data Mover: 2-D tile load Global -> LDS (bf16 elements).
// D# bit layout per CDNA5 ISA:
//  group0: [1:0]=count=1, [63:32]=lds_addr, [120:64]=global_addr,
//          [127:126]=type=2
//  group1: [17:16]=data_size=1 (2B), [79:48]=tensor_dim0, [111:80]=tensor_dim1,
//          [127:112]=tile_dim0, [143:128]=tile_dim1, [207:160]=dim0_stride
// This toolchain exposes the 6-arg builtin:
//  (uint32x4 g0, int32x8 g1, int32x4 g2, int32x4 g3, int32x8 g4, i32 cpol)
// ---------------------------------------------------------------------------
__device__ inline void tdm_load_2d_bf16(const bf16_t* gptr, unsigned lds_off,
                                        unsigned tile_d0, unsigned tile_d1,
                                        unsigned tensor_d0, unsigned tensor_d1,
                                        unsigned stride_d0) {
  unsigned long long ga = (unsigned long long)(uintptr_t)gptr;
  v4u g0;
  g0[0] = 1u;                                   // count=1, user mode
  g0[1] = lds_off;                              // lds_addr (bytes)
  g0[2] = (unsigned)ga;                         // global_addr[31:0]
  g0[3] = (unsigned)(ga >> 32) | (2u << 30);    // global_addr[56:32] | type=2
  v8i g1;
  g1[0] = (int)(1u << 16);                      // data_size = 2 bytes
  g1[1] = (int)((tensor_d0 & 0xFFFFu) << 16);   // tensor_dim0[15:0] at bit 48
  g1[2] = (int)((tensor_d0 >> 16) | ((tensor_d1 & 0xFFFFu) << 16));
  g1[3] = (int)((tensor_d1 >> 16) | (tile_d0 << 16));   // tile_dim0 at bit 112
  g1[4] = (int)tile_d1;                         // tile_dim1 at bit 128
  g1[5] = (int)stride_d0;                       // tensor_dim0_stride at bit 160
  g1[6] = 0;
  g1[7] = 0;
  v4i z4 = {0, 0, 0, 0};                        // groups 2/3 unused (2-D)
  v8i z8 = {0, 0, 0, 0, 0, 0, 0, 0};
  __builtin_amdgcn_tensor_load_to_lds(g0, g1, z4, z4, z8, /*cpol=*/0);
}

__device__ inline unsigned lds_offset_of(const void* p) {
  return (unsigned)(uintptr_t)(__attribute__((address_space(3))) const char*)p;
}

// ---------------------------------------------------------------------------
// Kernel 0: transpose+convert value: vT[b][v][qq] = bf16(value[b][qq][v])
// 32x32 LDS tile (padded, conflict-free), coalesced loads & packed stores.
// ---------------------------------------------------------------------------
__global__ __launch_bounds__(256) void transpose_value_kernel(
    const float* __restrict__ V, bf16_t* __restrict__ VT) {
  __shared__ float tile[32][33];
  int b  = blockIdx.z;
  int q0 = blockIdx.x * 32;   // qq tile origin (2048/32 = 64)
  int v0 = blockIdx.y * 32;   // v  tile origin (512/32  = 16)
  const float* Vb = V + (size_t)b * TT * DD;
  bf16_t* Tb = VT + (size_t)b * DD * TT;

  int tid = threadIdx.x;
  int r  = tid >> 3;          // 0..31  (qq row within tile)
  int c4 = (tid & 7) << 2;    // 0,4,...,28 (v col group)

  float4 ld = *(const float4*)(Vb + (size_t)(q0 + r) * DD + v0 + c4);
  tile[r][c4 + 0] = ld.x;
  tile[r][c4 + 1] = ld.y;
  tile[r][c4 + 2] = ld.z;
  tile[r][c4 + 3] = ld.w;
  __syncthreads();

  // write out: row = v, col = qq (4 consecutive bf16 packed into 8B store)
  v4bf o;
#pragma unroll
  for (int j = 0; j < 4; ++j) o[j] = (__bf16)tile[c4 + j][r];
  *(v4bf*)(Tb + (size_t)(v0 + r) * TT + q0 + c4) = o;
}

// ---------------------------------------------------------------------------
// Kernel 1: projection  P[t,i] = sum_d X[t,d] * W[i,d] + b[i]   (bf16 out)
// WG = 256 threads = 8 waves; each wave: 16(t) x 64(i); WG: 128(t) x 64(i).
// ---------------------------------------------------------------------------
__global__ __launch_bounds__(256) void proj_kernel(
    const float* __restrict__ X, const float* __restrict__ W,
    const float* __restrict__ bias, bf16_t* __restrict__ P) {
  int lane = threadIdx.x & 31;
  int wave = threadIdx.x >> 5;
  int t0 = blockIdx.x * 128 + wave * 16;
  int i0 = blockIdx.y * 64;

  v8f acc[4] = {};
  for (int k0 = 0; k0 < DD; k0 += 32) {
    v16bf a = load_afrag_f32(X + (size_t)t0 * DD + k0, DD, lane);
#pragma unroll
    for (int nt = 0; nt < 4; ++nt) {
      v16bf b = load_bfrag_rows_f32(W + (size_t)(i0 + nt * 16) * DD + k0, DD, lane);
      acc[nt] = wmma_bf16(a, b, acc[nt]);
    }
  }

  int n  = lane & 15;
  int mo = (lane >> 4) << 3;
#pragma unroll
  for (int nt = 0; nt < 4; ++nt) {
    int col = i0 + nt * 16 + n;
    float bv = bias[col];
#pragma unroll
    for (int v = 0; v < 8; ++v) {
      int row = t0 + mo + v;
      P[(size_t)row * DD + col] = (bf16_t)(acc[nt][v] + bv);
    }
  }
}

// ---------------------------------------------------------------------------
// Kernel 2: scores  S[b,kk,qq] = (1/T) * sum_i K[b,kk,i] * Q[b,qq,i]
// Per wave: 16(kk) x 64(qq); WG: 128(kk) x 64(qq); grid.z = batch.
// Q block (64 rows x 32 k, shared by all 8 waves) staged to LDS by the
// Tensor Data Mover, DOUBLE-BUFFERED: at step i wave0 issues the DMA for
// step i+1 into the alternate buffer, then waits TENSORcnt<=1 (TDM ops from
// one wave complete in order, so <=1 outstanding means tile i has landed).
// DMA of tile i+1 overlaps the WMMAs of tile i.
// ---------------------------------------------------------------------------
__global__ __launch_bounds__(256) void score_kernel(
    const bf16_t* __restrict__ Kp, const bf16_t* __restrict__ Qp,
    float* __restrict__ S) {
  __shared__ bf16_t qtile[2][64 * 32];   // 2 x 4 KB: rows = qq(64), cols = k(32)
  int lane = threadIdx.x & 31;
  int wave = threadIdx.x >> 5;
  int b   = blockIdx.z;
  int kk0 = blockIdx.x * 128 + wave * 16;
  int qq0 = blockIdx.y * 64;
  const bf16_t* Kb = Kp + (size_t)b * TT * DD;
  const bf16_t* Qrow = Qp + (size_t)b * TT * DD + (size_t)qq0 * DD;
  unsigned lds_buf[2] = { lds_offset_of(qtile[0]), lds_offset_of(qtile[1]) };

  constexpr int NSTEP = DD / 32;   // 16

  // prologue: DMA tile 0 into buffer 0
  if (wave == 0) {
    tdm_load_2d_bf16(Qrow, lds_buf[0], /*tile_d0=*/32, /*tile_d1=*/64,
                     /*tensor_d0=*/DD, /*tensor_d1=*/TT, /*stride_d0=*/DD);
  }

  v8f acc[4] = {};
  for (int it = 0; it < NSTEP; ++it) {
    int k0 = it * 32;
    if (wave == 0) {
      if (it + 1 < NSTEP) {
        // issue next tile into the other buffer (safe: barrier at the end of
        // the previous step guarantees all waves finished reading it)
        tdm_load_2d_bf16(Qrow + k0 + 32, lds_buf[(it + 1) & 1],
                         32, 64, DD, TT, DD);
        __builtin_amdgcn_s_wait_tensorcnt(1);   // current tile has landed
      } else {
        __builtin_amdgcn_s_wait_tensorcnt(0);   // last tile
      }
    }
    __syncthreads();                            // current buffer visible to all

    const bf16_t* buf = qtile[it & 1];
    v16bf a = load_afrag_bf16(Kb + (size_t)kk0 * DD + k0, DD, lane);
#pragma unroll
    for (int nt = 0; nt < 4; ++nt) {
      // B-frag: lane n=lane&15 of column block nt, k-run (lane&16)+e
      const bf16_t* lp = buf + ((nt * 16 + (lane & 15)) * 32 + (lane & 16));
      v16bf bq = *(const v16bf*)lp;             // 32B contiguous LDS read
      acc[nt] = wmma_bf16(a, bq, acc[nt]);
    }
    __syncthreads();                            // done reading current buffer
  }

  float* Sb = S + (size_t)b * TT * TT;
  int n  = lane & 15;
  int mo = (lane >> 4) << 3;
#pragma unroll
  for (int nt = 0; nt < 4; ++nt) {
    int col = qq0 + nt * 16 + n;
#pragma unroll
    for (int v = 0; v < 8; ++v) {
      int row = kk0 + mo + v;
      Sb[(size_t)row * TT + col] = acc[nt][v] * INV_TEMP;
    }
  }
}

// ---------------------------------------------------------------------------
// Kernel 3: in-place masked softmax over the last axis (TQ = 2048).
// One 256-thread block per row; wave shuffle + LDS cross-wave reduction.
// ---------------------------------------------------------------------------
__global__ __launch_bounds__(256) void softmax_kernel(
    float* __restrict__ S, const unsigned char* __restrict__ mask) {
  size_t row = blockIdx.x;                 // 0 .. NB*TT-1
  float* p = S + row * TT;
  const unsigned char* mrow = mask + row * TT;
  int tid = threadIdx.x;

  float x[8];
#pragma unroll
  for (int j = 0; j < 8; ++j) {
    int c = tid + j * 256;
    float v = p[c];
    if (mrow[c]) v = -INFINITY;
    x[j] = v;
  }

  __shared__ float red[8];
  int wv = tid >> 5;

  float m = x[0];
#pragma unroll
  for (int j = 1; j < 8; ++j) m = fmaxf(m, x[j]);
#pragma unroll
  for (int off = 16; off > 0; off >>= 1) m = fmaxf(m, __shfl_xor(m, off, 32));
  if ((tid & 31) == 0) red[wv] = m;
  __syncthreads();
  float mall = red[0];
#pragma unroll
  for (int w = 1; w < 8; ++w) mall = fmaxf(mall, red[w]);
  __syncthreads();

  float s = 0.0f;
#pragma unroll
  for (int j = 0; j < 8; ++j) { x[j] = __expf(x[j] - mall); s += x[j]; }
#pragma unroll
  for (int off = 16; off > 0; off >>= 1) s += __shfl_xor(s, off, 32);
  if ((tid & 31) == 0) red[wv] = s;
  __syncthreads();
  float sall = 0.0f;
#pragma unroll
  for (int w = 0; w < 8; ++w) sall += red[w];

  float inv = 1.0f / sall;
#pragma unroll
  for (int j = 0; j < 8; ++j) p[tid + j * 256] = x[j] * inv;
}

// ---------------------------------------------------------------------------
// Kernel 4: out[b,kk,v] = sum_qq attn[b,kk,qq] * vT[b,v,qq]
// Waves arranged 2(m) x 4(n); per wave 16(kk) x 64(v); WG: 32(kk) x 256(v).
// ---------------------------------------------------------------------------
__global__ __launch_bounds__(256) void av_kernel(
    const float* __restrict__ S, const bf16_t* __restrict__ VT,
    float* __restrict__ O) {
  int lane = threadIdx.x & 31;
  int wave = threadIdx.x >> 5;
  int wm = wave & 1;
  int wn = wave >> 1;
  int b   = blockIdx.z;
  int kk0 = blockIdx.x * 32 + wm * 16;
  int vv0 = blockIdx.y * 256 + wn * 64;
  const float*  Sb = S  + (size_t)b * TT * TT;
  const bf16_t* Tb = VT + (size_t)b * DD * TT;

  v8f acc[4] = {};
  for (int q0 = 0; q0 < TT; q0 += 32) {
    v16bf a = load_afrag_f32(Sb + (size_t)kk0 * TT + q0, TT, lane);
#pragma unroll
    for (int nt = 0; nt < 4; ++nt) {
      v16bf bv = load_bfrag_rows_bf16(Tb + (size_t)(vv0 + nt * 16) * TT + q0, TT, lane);
      acc[nt] = wmma_bf16(a, bv, acc[nt]);
    }
  }

  float* Ob = O + (size_t)b * TT * DD;
  int n  = lane & 15;
  int mo = (lane >> 4) << 3;
#pragma unroll
  for (int nt = 0; nt < 4; ++nt) {
    int col = vv0 + nt * 16 + n;
#pragma unroll
    for (int v = 0; v < 8; ++v) {
      int row = kk0 + mo + v;
      Ob[(size_t)row * DD + col] = acc[nt][v];
    }
  }
}

// ---------------------------------------------------------------------------
// Host-side launcher
// ---------------------------------------------------------------------------
extern "C" void kernel_launch(void* const* d_in, const int* in_sizes, int n_in,
                              void* d_out, int out_size, void* d_ws,
                              size_t ws_size, hipStream_t stream) {
  const float*         query = (const float*)d_in[0];
  const float*         key   = (const float*)d_in[1];
  const float*         value = (const float*)d_in[2];
  const unsigned char* mask  = (const unsigned char*)d_in[3];
  const float*         Wq    = (const float*)d_in[4];
  const float*         bq    = (const float*)d_in[5];
  const float*         Wk    = (const float*)d_in[6];
  const float*         bk    = (const float*)d_in[7];

  float* attn = (float*)d_out;                       // [NB, TT, TT]
  float* outp = attn + (size_t)NB * TT * TT;         // [NB, TT, DD]

  bf16_t* qbf = (bf16_t*)d_ws;                       // [NB*TT, DD] bf16
  bf16_t* kbf = qbf + (size_t)NB * TT * DD;          // [NB*TT, DD] bf16
  bf16_t* vbT = kbf + (size_t)NB * TT * DD;          // [NB, DD, TT] bf16

  dim3 blk(256);
  const int NT = NB * TT;  // 16384 projected rows

  transpose_value_kernel<<<dim3(TT / 32, DD / 32, NB), blk, 0, stream>>>(value, vbT);
  proj_kernel<<<dim3(NT / 128, DD / 64), blk, 0, stream>>>(query, Wq, bq, qbf);
  proj_kernel<<<dim3(NT / 128, DD / 64), blk, 0, stream>>>(key, Wk, bk, kbf);
  score_kernel<<<dim3(TT / 128, TT / 64, NB), blk, 0, stream>>>(kbf, qbf, attn);
  softmax_kernel<<<dim3(NB * TT), blk, 0, stream>>>(attn, mask);
  av_kernel<<<dim3(TT / 32, DD / 256, NB), blk, 0, stream>>>(attn, vbT, outp);
}
